// NTXentLoss_46514495815716
// MI455X (gfx1250) — matrix operand order repaired and checked by reading, not compiled, not run on hardware
//
#include <hip/hip_runtime.h>
#include <hip/hip_bf16.h>

typedef __attribute__((ext_vector_type(16))) _Float16     v16h;
typedef __attribute__((ext_vector_type(8)))  _Float16     v8h;
typedef __attribute__((ext_vector_type(4)))  _Float16     v4h;
typedef __attribute__((ext_vector_type(8)))  float        v8f;
typedef __attribute__((ext_vector_type(4)))  unsigned int v4u;
typedef __attribute__((ext_vector_type(8)))  int          v8i;
typedef __attribute__((ext_vector_type(4)))  int          v4i;

static constexpr int   Bn      = 2048;
static constexpr int   Dd      = 512;
static constexpr int   Nn      = 4096;          // 2*Bn
static constexpr float TEMPINV = 2.0f;          // 1 / 0.5
static constexpr float EPSN    = 1e-8f;

// ---------------------------------------------------------------------------
// TDM: DMA one 16-row x 512-col f16 tile (16 KB) from global into LDS.
// 2-D descriptor: data_size=2B, tile_dim0=512 (contiguous), tile_dim1=16,
// tensor_dim0_stride=512. Groups 2/3 unused (tile_dim2=0 -> 2-D).
// ---------------------------------------------------------------------------
__device__ __forceinline__ void tdm_load_tile16(const _Float16* gsrc,
                                                unsigned lds_off) {
  union { unsigned u[4]; v4u v; } g0;
  union { int      i[8]; v8i v; } g1;
  const unsigned long long ga = (unsigned long long)gsrc;

  g0.u[0] = 1u;                                      // count=1, user mode
  g0.u[1] = lds_off;                                 // lds_addr (bytes)
  g0.u[2] = (unsigned)(ga & 0xFFFFFFFFu);            // global_addr[31:0]
  g0.u[3] = (unsigned)((ga >> 32) & 0x01FFFFFFu)     // global_addr[56:32]
          | 0x80000000u;                             // type=2 ("image")

  g1.i[0] = 0x00010000;                              // data_size=1 (2 bytes)
  g1.i[1] = (int)((unsigned)Dd << 16);               // tensor_dim0 = 512
  g1.i[2] = (int)((unsigned)Nn << 16);               // tensor_dim1 = 4096
  g1.i[3] = (int)((unsigned)Dd << 16);               // tile_dim0   = 512
  g1.i[4] = 16;                                      // tile_dim1=16, tile_dim2=0
  g1.i[5] = Dd;                                      // tensor_dim0_stride = 512
  g1.i[6] = 0;
  g1.i[7] = 0;

  v4i z4 = {};
#if __has_include(<hip/amd_detail/amd_gfx1250_TDM.h>)
  v8i z8 = {};
  __builtin_amdgcn_tensor_load_to_lds(g0.v, g1.v, z4, z4, z8, 0);
#else
  __builtin_amdgcn_tensor_load_to_lds(g0.v, g1.v, z4, z4, 0);
#endif
}

// ---------------------------------------------------------------------------
// Kernel 1: row-normalize concat(z_i, z_j) -> f16 zn[N, D]; also zero S.
// One wave per row.
// ---------------------------------------------------------------------------
__global__ void __launch_bounds__(256)
nt_normalize(const float* __restrict__ zi, const float* __restrict__ zj,
             _Float16* __restrict__ zn, float* __restrict__ S) {
  const int wave = threadIdx.x >> 5;
  const int lane = threadIdx.x & 31;
  const int row  = blockIdx.x * 8 + wave;
  if (row >= Nn) return;

  const float* src = (row < Bn) ? (zi + (size_t)row * Dd)
                                : (zj + (size_t)(row - Bn) * Dd);
  float vals[16];
  float ss = 0.0f;
#pragma unroll
  for (int i = 0; i < 4; ++i) {
    const float4 v = ((const float4*)src)[lane + i * 32];
    vals[i * 4 + 0] = v.x; vals[i * 4 + 1] = v.y;
    vals[i * 4 + 2] = v.z; vals[i * 4 + 3] = v.w;
    ss += v.x * v.x + v.y * v.y + v.z * v.z + v.w * v.w;
  }
#pragma unroll
  for (int m = 16; m >= 1; m >>= 1) ss += __shfl_xor(ss, m, 32);

  const float inv = 1.0f / fmaxf(sqrtf(ss), EPSN);

  _Float16* dst = zn + (size_t)row * Dd;
#pragma unroll
  for (int i = 0; i < 4; ++i) {
    v4h h;
    h.x = (_Float16)(vals[i * 4 + 0] * inv);
    h.y = (_Float16)(vals[i * 4 + 1] * inv);
    h.z = (_Float16)(vals[i * 4 + 2] * inv);
    h.w = (_Float16)(vals[i * 4 + 3] * inv);
    ((v4h*)dst)[lane + i * 32] = h;
  }
  if (lane == 0) S[row] = 0.0f;      // for atomic accumulation in kernel 2
}

// ---------------------------------------------------------------------------
// Kernel 2: fused sim-GEMM + masked sum-exp + pos extraction.
// Grid (16,16): blockIdx.y -> 256-row block, blockIdx.x -> 256-col strip.
// 8 waves; wave w owns rows m0+32w .. +31 (two 16-row A tiles, regs).
// B tiles (16 cols x K=512, 16 KB) stream through double-buffered LDS via
// the Tensor Data Mover, overlapped with the 32 WMMAs of the previous tile.
// The whole B tile is preloaded LDS->VGPR (128 VGPRs) so all ds_loads issue
// as one clause and the 32 WMMAs run back-to-back behind a single wait.
// ---------------------------------------------------------------------------
__global__ void __launch_bounds__(256)
__attribute__((amdgpu_waves_per_eu(1)))
nt_simexp(const _Float16* __restrict__ zn,
          float* __restrict__ S, float* __restrict__ pos) {
  __shared__ alignas(64) _Float16 tileB[2][16 * Dd];   // 2 x 16 KB

  const int wave = threadIdx.x >> 5;
  const int lane = threadIdx.x & 31;
  const int half = lane >> 4;        // 0: lanes 0-15, 1: lanes 16-31
  const int l15  = lane & 15;
  const int mrow = blockIdx.y * 256 + wave * 32;   // this wave's first row
  const int ncol = blockIdx.x * 256;               // strip's first column
  constexpr int TILES = 16;                        // 16 col tiles of 16

  // --- Register-resident A: 32 rows x K=512 as two 16-row fragment sets.
  // A k order used here (contiguous, same as B) per 16x32 chunk:
  //   lanes 0-15: M=lane,    V0-7: K=kb+0..15
  //   lanes16-31: M=lane-16, V0-7: K=kb+16..31
  v16h afrag0[16], afrag1[16];
  {
    const _Float16* a0 = zn + (size_t)(mrow + l15) * Dd;
    const _Float16* a1 = a0 + (size_t)16 * Dd;
#pragma unroll
    for (int kc = 0; kc < 16; ++kc) {
      const int kb = kc * 32;
      v8h lo = *(const v8h*)(a0 + kb + half * 16);
      v8h hi = *(const v8h*)(a0 + kb + 8 + half * 16);
      afrag0[kc] = __builtin_shufflevector(lo, hi,
          0, 1, 2, 3, 4, 5, 6, 7, 8, 9, 10, 11, 12, 13, 14, 15);
      lo = *(const v8h*)(a1 + kb + half * 16);
      hi = *(const v8h*)(a1 + kb + 8 + half * 16);
      afrag1[kc] = __builtin_shufflevector(lo, hi,
          0, 1, 2, 3, 4, 5, 6, 7, 8, 9, 10, 11, 12, 13, 14, 15);
    }
  }

  float acc0[8], acc1[8];
#pragma unroll
  for (int r = 0; r < 8; ++r) { acc0[r] = 0.0f; acc1[r] = 0.0f; }

  // Prologue: DMA tile 0 into buffer 0.
  if (wave == 0) {
    tdm_load_tile16(zn + (size_t)ncol * Dd, (unsigned)(size_t)&tileB[0][0]);
    __builtin_amdgcn_s_wait_tensorcnt(0);
  }
  __syncthreads();

  for (int t = 0; t < TILES; ++t) {
    // Prefetch next B tile into the other buffer while we compute.
    if (wave == 0 && t + 1 < TILES)
      tdm_load_tile16(zn + (size_t)(ncol + (t + 1) * 16) * Dd,
                      (unsigned)(size_t)&tileB[(t + 1) & 1][0]);

    // B layout (32x16, K x N): lanes 0-15: N=lane, K=kb+0..15 contiguous;
    // lanes 16-31: N=lane-16, K=kb+16..31. B[k,n] = tile row n, col k.
    const _Float16* blane = &tileB[t & 1][0] + (size_t)l15 * Dd + half * 16;

    // Preload the full tile's B fragments (16 x v16h = 128 VGPRs): all 32
    // ds_load_b128 issue up front, then the WMMAs drain behind one wait.
    v16h bfr[16];
#pragma unroll
    for (int kc = 0; kc < 16; ++kc)
      bfr[kc] = *(const v16h*)(blane + kc * 32);

    v8f c0 = {}, c1 = {};
#pragma unroll
    for (int kc = 0; kc < 16; ++kc) {
      c0 = __builtin_amdgcn_wmma_f32_16x16x32_f16(
          false, afrag0[kc], false, bfr[kc], (short)0, c0, false, false);
      c1 = __builtin_amdgcn_wmma_f32_16x16x32_f16(
          false, afrag1[kc], false, bfr[kc], (short)0, c1, false, false);
    }

    // Epilogue: C layout — VGPR r, lane l => (M = r + 8*(l>>4), N = l&15).
    const int gN = ncol + t * 16 + l15;
#pragma unroll
    for (int r = 0; r < 8; ++r) {
      const int gM0 = mrow + r + 8 * half;
      const int gM1 = gM0 + 16;
      const float v0 = c0[r] * TEMPINV;
      const float v1 = c1[r] * TEMPINV;
      if (gN == gM0 + Bn || gN + Bn == gM0) pos[gM0] = v0;   // positive pair
      if (gN == gM1 + Bn || gN + Bn == gM1) pos[gM1] = v1;
      acc0[r] += (gM0 == gN) ? 0.0f : __expf(v0);            // diag masked
      acc1[r] += (gM1 == gN) ? 0.0f : __expf(v1);
    }

    if (wave == 0) __builtin_amdgcn_s_wait_tensorcnt(0);
    __syncthreads();   // next buffer ready; all reads of current buffer done
  }

  // Per-row sums: reduce across the 16 lanes of each half-wave, then
  // atomically merge this column-strip's partial into S[row].
#pragma unroll
  for (int r = 0; r < 8; ++r) {
    float a = acc0[r];
    float b = acc1[r];
#pragma unroll
    for (int m = 8; m >= 1; m >>= 1) {
      a += __shfl_xor(a, m, 16);
      b += __shfl_xor(b, m, 16);
    }
    if (l15 == r) {
      atomicAdd(&S[mrow + r + 8 * half], a);
      atomicAdd(&S[mrow + 16 + r + 8 * half], b);
    }
  }
}

// ---------------------------------------------------------------------------
// Kernel 3: loss = mean_i( log(S[i] + exp(pos[i])) - pos[i] )
// ---------------------------------------------------------------------------
__global__ void __launch_bounds__(256)
nt_finalize(const float* __restrict__ S, const float* __restrict__ pos,
            float* __restrict__ out) {
  __shared__ float red[256];
  float s = 0.0f;
  for (int r = threadIdx.x; r < Nn; r += 256) {
    const float p = pos[r];
    s += logf(S[r] + expf(p)) - p;
  }
  red[threadIdx.x] = s;
  __syncthreads();
#pragma unroll
  for (int st = 128; st > 0; st >>= 1) {
    if (threadIdx.x < st) red[threadIdx.x] += red[threadIdx.x + st];
    __syncthreads();
  }
  if (threadIdx.x == 0) out[0] = red[0] / (float)Nn;
}

// ---------------------------------------------------------------------------
extern "C" void kernel_launch(void* const* d_in, const int* in_sizes, int n_in,
                              void* d_out, int out_size, void* d_ws, size_t ws_size,
                              hipStream_t stream) {
  (void)in_sizes; (void)n_in; (void)out_size; (void)ws_size;
  const float* zi = (const float*)d_in[0];
  const float* zj = (const float*)d_in[1];
  // d_in[2] (labels) is ignored by the reference.

  _Float16* zn = (_Float16*)d_ws;                                   // 4 MB
  float* S     = (float*)((char*)d_ws + (size_t)Nn * Dd * sizeof(_Float16));
  float* pos   = S + Nn;

  nt_normalize<<<Nn / 8, 256, 0, stream>>>(zi, zj, zn, S);
  nt_simexp  <<<dim3(16, 16), 256, 0, stream>>>(zn, S, pos);
  nt_finalize<<<1, 256, 0, stream>>>(S, pos, (float*)d_out);
}
